// ConvPair_55198919688509
// MI455X (gfx1250) — matrix operand order repaired and checked by reading, not compiled, not run on hardware
//
#include <hip/hip_runtime.h>

#define Nn 1024
#define Ff 8
#define Hh 128
#define Oo 3

typedef __attribute__((ext_vector_type(16))) _Float16 v16h;
typedef __attribute__((ext_vector_type(8)))  float    v8f;
typedef __attribute__((ext_vector_type(4)))  float    v4f;

__device__ __forceinline__ float fast_tanh(float x) {
#if __has_builtin(__builtin_amdgcn_tanhf)
  return __builtin_amdgcn_tanhf(x);          // v_tanh_f32 (gfx1250 trans op)
#else
  float e = __builtin_amdgcn_exp2f(x * 2.88539008177792681f);
  return 1.0f - 2.0f / (e + 1.0f);
#endif
}

// XOR-butterfly add across the 16-lane row using DPP16 row_xmask (VALU, not DS pipe).
template <int MASK>
__device__ __forceinline__ float xmask_add(float v) {
#if __has_builtin(__builtin_amdgcn_update_dpp)
  int p = __builtin_amdgcn_update_dpp(0, __float_as_int(v),
                                      0x160 | MASK, 0xF, 0xF, true);
  return v + __int_as_float(p);
#else
  return v + __shfl_xor(v, MASK);
#endif
}

// Stage 1: a[i][k] = (x @ Wa)[i][k] + b1[k];  hj[j][k] = (x @ Wb)[j][k]
__global__ void pairmlp_prep(const float* __restrict__ x,
                             const float* __restrict__ W1,
                             const float* __restrict__ b1,
                             float* __restrict__ a,
                             float* __restrict__ hjw) {
  int idx = blockIdx.x * blockDim.x + threadIdx.x;   // 0 .. N*H-1
  int row = idx >> 7;
  int k   = idx & (Hh - 1);
  const float* xr = x + row * Ff;
  float sa = 0.f, sb = 0.f;
#pragma unroll
  for (int f = 0; f < Ff; ++f) {
    float xv = xr[f];
    sa = fmaf(xv, W1[f * Hh + k], sa);
    sb = fmaf(xv, W1[(Ff + f) * Hh + k], sb);
  }
  a[idx]   = sa + b1[k];
  hjw[idx] = sb;
}

// Stage 2+3: per wave, one output row i. Loop j in chunks of 16:
//   A(16x128 f16) = tanh(a[i] + hj[j0..j0+15])  -> wmma vs register/LDS-resident W2 (f16)
//   epilogue: tanh(C + b2) . W3, tanh(+b3), reduce over (j, o) into out[i].
__global__ __launch_bounds__(256) void pairmlp_main(
    const float* __restrict__ a, const float* __restrict__ hjw,
    const float* __restrict__ W2, const float* __restrict__ b2,
    const float* __restrict__ W3, const float* __restrict__ b3,
    float* __restrict__ out) {
  // W2 pre-swizzled into B-fragment order:
  // flat = ((ntile*4 + kstep)*32 + lane)*16 + h
  //   k = kstep*32 + (h/8)*16 + (lane/16)*8 + h%8 ; n = ntile*16 + (lane%16)
  __shared__ __align__(32) _Float16 ldsB[8 * 4 * 32 * 16];   // 32 KB

  const int t = threadIdx.x;
#pragma unroll 4
  for (int u = 0; u < 64; ++u) {
    int flat  = u * 256 + t;
    int h     = flat & 15;
    int lane_ = (flat >> 4) & 31;
    int ks    = (flat >> 9) & 3;
    int nt    = flat >> 11;
    int k     = ks * 32 + (h >> 3) * 16 + (lane_ >> 4) * 8 + (h & 7);
    int n     = nt * 16 + (lane_ & 15);
    ldsB[flat] = (_Float16)W2[k * Hh + n];
  }
  __syncthreads();

  const int wave = t >> 5;
  const int lane = t & 31;
  const int half = lane >> 4;     // 0: rows 0-7 / K lo-group ; 1: rows 8-15 / K hi-group
  const int m16  = lane & 15;

  const int i = blockIdx.x * 8 + wave;
  const float* arow = a + i * Hh;

  // loop-invariant per-lane caches
  float b2v[8], w3c[8][3], b3v[3];
#pragma unroll
  for (int nt = 0; nt < 8; ++nt) {
    int n = nt * 16 + m16;
    b2v[nt] = b2[n];
    w3c[nt][0] = W3[n * Oo + 0];
    w3c[nt][1] = W3[n * Oo + 1];
    w3c[nt][2] = W3[n * Oo + 2];
  }
  b3v[0] = b3[0]; b3v[1] = b3[1]; b3v[2] = b3[2];

  const v16h* Bp = (const v16h*)ldsB;
  float acc = 0.f;

  for (int j0 = 0; j0 < Nn; j0 += 16) {
    const float* hrow = hjw + (j0 + m16) * Hh;   // this lane's A row: j = j0 + (lane%16)

    // Build A fragments (16-bit A 16x32 layout): lane holds row m=lane%16,
    // halves 0..7 -> k = kb..kb+7, halves 8..15 -> k = kb+16..kb+23, kb = ks*32 + half*8
    v16h Af[4];
#pragma unroll
    for (int ks = 0; ks < 4; ++ks) {
      int kb = ks * 32 + half * 8;
      const v4f* ap = (const v4f*)(arow + kb);
      const v4f* hp = (const v4f*)(hrow + kb);
      v4f a0 = ap[0], a1 = ap[1], a2 = ap[4], a3 = ap[5];
      v4f h0 = hp[0], h1 = hp[1], h2 = hp[4], h3 = hp[5];
#pragma unroll
      for (int e = 0; e < 4; ++e) {
        Af[ks][e]      = (_Float16)fast_tanh(a0[e] + h0[e]);
        Af[ks][4 + e]  = (_Float16)fast_tanh(a1[e] + h1[e]);
        Af[ks][8 + e]  = (_Float16)fast_tanh(a2[e] + h2[e]);
        Af[ks][12 + e] = (_Float16)fast_tanh(a3[e] + h3[e]);
      }
    }

    // GEMM: C[16x128] = A @ W2 ; 8 N-tiles x 4 K-steps = 32 WMMAs
    v8f Cacc[8] = {};
#pragma unroll
    for (int ks = 0; ks < 4; ++ks) {
#pragma unroll
      for (int nt = 0; nt < 8; ++nt) {
        v16h Bf = Bp[(nt * 4 + ks) * 32 + lane];
        Cacc[nt] = __builtin_amdgcn_wmma_f32_16x16x32_f16(
            false, Af[ks], false, Bf, (short)0, Cacc[nt], false, false);
      }
    }

    // Epilogue: th = tanh(C + b2); per-row dot with W3 (partial over lane's 8 cols)
    float part[8][3] = {};
#pragma unroll
    for (int nt = 0; nt < 8; ++nt) {
#pragma unroll
      for (int c = 0; c < 8; ++c) {
        float th = fast_tanh(Cacc[nt][c] + b2v[nt]);
        part[c][0] = fmaf(th, w3c[nt][0], part[c][0]);
        part[c][1] = fmaf(th, w3c[nt][1], part[c][1]);
        part[c][2] = fmaf(th, w3c[nt][2], part[c][2]);
      }
    }
    // cross-lane dot reduction within each 16-lane half via DPP row_xmask adds
    // (row m = c + 8*half lives across lanes of that half)
    float cv = 0.f;
#pragma unroll
    for (int c = 0; c < 8; ++c) {
#pragma unroll
      for (int o = 0; o < 3; ++o) {
        float v = part[c][o];
        v = xmask_add<1>(v);
        v = xmask_add<2>(v);
        v = xmask_add<4>(v);
        v = xmask_add<8>(v);
        cv += fast_tanh(v + b3v[o]);
      }
    }
    acc += (m16 == 0) ? cv : 0.0f;   // rows 0-7 live in lane0, rows 8-15 in lane16
  }

  acc += __shfl_xor(acc, 16);        // combine the two halves (once per wave)
  if (lane == 0) out[i] = acc;
}

extern "C" void kernel_launch(void* const* d_in, const int* in_sizes, int n_in,
                              void* d_out, int out_size, void* d_ws, size_t ws_size,
                              hipStream_t stream) {
  const float* x  = (const float*)d_in[0];
  const float* W1 = (const float*)d_in[1];
  const float* b1 = (const float*)d_in[2];
  const float* W2 = (const float*)d_in[3];
  const float* b2 = (const float*)d_in[4];
  const float* W3 = (const float*)d_in[5];
  const float* b3 = (const float*)d_in[6];
  float* outp = (float*)d_out;

  float* a   = (float*)d_ws;            // N*H f32 : hi + b1
  float* hjw = a + Nn * Hh;             // N*H f32 : hj
  (void)in_sizes; (void)n_in; (void)out_size; (void)ws_size;

  pairmlp_prep<<<(Nn * Hh) / 256, 256, 0, stream>>>(x, W1, b1, a, hjw);
  pairmlp_main<<<Nn / 8, 256, 0, stream>>>(a, hjw, W2, b2, W3, b3, outp);
}
